// Geo2Vec_Model_37512244363597
// MI455X (gfx1250) — matrix operand order repaired and checked by reading, not compiled, not run on hardware
//
#include <hip/hip_runtime.h>

// ---- types for WMMA ----
typedef __attribute__((ext_vector_type(16))) __bf16 v16bf;
typedef __attribute__((ext_vector_type(8)))  __bf16 v8bf;
typedef __attribute__((ext_vector_type(8)))  float  v8f;

#define HID    256
#define IN1    320
#define INMID  576
#define MTILE  128
#define NLAYER 8
#define SLOPE  0.01f

// ---------------------------------------------------------------------------
// Prep: transpose + fp32->bf16 convert weights into workspace.
// in:  [L][K][N] fp32   out: [L][N][K] bf16
// ---------------------------------------------------------------------------
__global__ void wt_transpose_kernel(const float* __restrict__ in,
                                    __bf16* __restrict__ out,
                                    int L, int K, int N) {
    long total = (long)L * K * N;
    for (long e = (long)blockIdx.x * blockDim.x + threadIdx.x; e < total;
         e += (long)gridDim.x * blockDim.x) {
        long l = e / ((long)K * N);
        long r = e % ((long)K * N);
        long n = r / K;
        long k = r % K;
        out[e] = (__bf16)in[(l * K + k) * N + n];
    }
}

// ---------------------------------------------------------------------------
// Block-level GEMM, N-sliced across waves:
//   wave w owns output columns [w*32, w*32+32) (two 16-wide N-tiles) and
//   iterates over all 8 M-stripes (128 rows). Each B fragment is loaded from
//   global (L2-resident) exactly once per block, double-buffered in registers,
//   and reused for 8 WMMAs into 8 independent accumulators.
//
//   kt / ntl loops are kept rolled (#pragma unroll 1) so live VGPRs stay
//   ~100 (64 acc + 2x8 B + in-flight A) and the compiler cannot pre-load all
//   A-fragments and spill to scratch (round-2 failure mode).
// ---------------------------------------------------------------------------
__device__ __forceinline__ void block_gemm(const __bf16* A, int sA, int K,
                                           const __bf16* __restrict__ WT,
                                           const float* __restrict__ bias,
                                           bool leaky,
                                           __bf16* D, int sD, int dcol0) {
    const int t    = threadIdx.x;
    const int wave = t >> 5;
    const int lane = t & 31;
    const int half = lane >> 4;
    const int l15  = lane & 15;
    const int KT   = K >> 5;

    const __bf16* Abase = A + l15 * sA + half * 16;

#pragma unroll 1
    for (int ntl = 0; ntl < 2; ++ntl) {
        const int nt = wave * 2 + ntl;
        const float bval = bias[nt * 16 + l15];

        v8f acc[8];
#pragma unroll
        for (int m = 0; m < 8; ++m)
#pragma unroll
            for (int v = 0; v < 8; ++v) acc[m][v] = bval;

        const __bf16* Wrow = WT + (long)(nt * 16 + l15) * K + half * 16;

        v16bf bcur = *(const v16bf*)(Wrow);
#pragma unroll 1
        for (int kt = 0; kt < KT; ++kt) {
            // prefetch next B fragment (branchless; last iter reloads same)
            const int kn = (kt + 1 < KT) ? (kt + 1) : kt;
            v16bf bnext = *(const v16bf*)(Wrow + kn * 32);
#pragma unroll
            for (int m = 0; m < 8; ++m) {
                const __bf16* ap = Abase + m * 16 * sA + kt * 32;
                v8bf alo = *(const v8bf*)(ap);        // K = kb + half*16 + 0..7
                v8bf ahi = *(const v8bf*)(ap + 8);    // K = kb + half*16 + 8..15
                v16bf a = __builtin_shufflevector(alo, ahi,
                            0,1,2,3,4,5,6,7,8,9,10,11,12,13,14,15);
                acc[m] = __builtin_amdgcn_wmma_f32_16x16x32_bf16(
                             false, a, false, bcur, (short)0, acc[m], false, false);
            }
            bcur = bnext;
        }

#pragma unroll
        for (int m = 0; m < 8; ++m)
#pragma unroll
            for (int v = 0; v < 8; ++v) {
                float x = acc[m][v];
                if (leaky) x = (x > 0.0f) ? x : SLOPE * x;
                const int row = m * 16 + v + 8 * half;
                D[row * sD + dcol0 + nt * 16 + l15] = (__bf16)x;
            }
    }
}

// ---------------------------------------------------------------------------
// Main fused MLP kernel: one block = 128 batch rows, 8 waves.
// ---------------------------------------------------------------------------
__global__ __launch_bounds__(256) void geo2vec_mlp_kernel(
        const float* __restrict__ xy,   const int* __restrict__ idx,
        const float* __restrict__ emb,
        const float* __restrict__ b1a,  const float* __restrict__ b1b,
        const float* __restrict__ ba,   const float* __restrict__ bb,
        const float* __restrict__ W2,   const float* __restrict__ b2,
        const __bf16* __restrict__ W1aT, const __bf16* __restrict__ W1bT,
        const __bf16* __restrict__ WaT,  const __bf16* __restrict__ WbT,
        float* __restrict__ out) {
    __shared__ __bf16 hbuf[MTILE * INMID];   // [posenc(64) | z(256) | x(256)]
    __shared__ __bf16 tbuf[MTILE * HID];     // mid activation

    const int t      = threadIdx.x;
    const int block0 = blockIdx.x * MTILE;

    // ---- positional encoding: one row per thread (threads 0..127) ----
    if (t < MTILE) {
        const int row = block0 + t;
        const float x = xy[2 * row + 0];
        const float y = xy[2 * row + 1];
        const float r = sqrtf(x * x + y * y);
        __bf16* f = &hbuf[t * INMID];
        f[0] = (__bf16)x;
        f[1] = (__bf16)y;
        f[42] = (__bf16)x;
        f[43] = (__bf16)y;
#pragma unroll
        for (int j = 0; j < 10; ++j) {
            const float fr = 1.0f + 5.0f * (float)j * (1.0f / 9.0f);
            f[2 + 2 * j]  = (__bf16)sinf(x * fr);
            f[3 + 2 * j]  = (__bf16)sinf(y * fr);
            f[22 + 2 * j] = (__bf16)cosf(x * fr);
            f[23 + 2 * j] = (__bf16)cosf(y * fr);
            f[44 + j]     = (__bf16)sinf(r * fr);
            f[54 + j]     = (__bf16)cosf(r * fr);
        }
    }
    // ---- embedding gather: 128 rows x 256 cols, all 256 threads ----
    for (int e = t; e < MTILE * HID; e += 256) {
        const int row = e >> 8;
        const int c   = e & 255;
        const int id  = idx[block0 + row];
        hbuf[row * INMID + 64 + c] = (__bf16)emb[(long)id * HID + c];
    }
    __syncthreads();

    // ---- layer 0: x = lrelu(xy_z @ W1a + b1a) @ W1b + b1b ----
    block_gemm(hbuf, INMID, IN1, W1aT, b1a, true,  tbuf, HID,  0);
    __syncthreads();
    block_gemm(tbuf, HID,   HID, W1bT, b1b, false, hbuf, INMID, IN1);
    __syncthreads();

    // ---- 8 residual-concat layers ----
    for (int i = 0; i < NLAYER; ++i) {
        block_gemm(hbuf, INMID, INMID, WaT + (long)i * HID * INMID,
                   ba + i * HID, true,  tbuf, HID, 0);
        __syncthreads();
        block_gemm(tbuf, HID, HID, WbT + (long)i * HID * HID,
                   bb + i * HID, false, hbuf, INMID, IN1);
        __syncthreads();
    }

    // ---- head: out = x @ W2 + b2 (256-dot per row) ----
    if (t < MTILE) {
        float acc = b2[0];
        const __bf16* xr = &hbuf[t * INMID + IN1];
#pragma unroll 8
        for (int c = 0; c < HID; ++c) acc += (float)xr[c] * W2[c];
        out[block0 + t] = acc;
    }
}

// ---------------------------------------------------------------------------
extern "C" void kernel_launch(void* const* d_in, const int* in_sizes, int n_in,
                              void* d_out, int out_size, void* d_ws, size_t ws_size,
                              hipStream_t stream) {
    const float* xy  = (const float*)d_in[0];
    const int*   idx = (const int*)d_in[1];
    const float* emb = (const float*)d_in[2];
    const float* W1a = (const float*)d_in[3];
    const float* b1a = (const float*)d_in[4];
    const float* W1b = (const float*)d_in[5];
    const float* b1b = (const float*)d_in[6];
    const float* Wa  = (const float*)d_in[7];
    const float* ba  = (const float*)d_in[8];
    const float* Wb  = (const float*)d_in[9];
    const float* bb  = (const float*)d_in[10];
    const float* W2  = (const float*)d_in[11];
    const float* b2  = (const float*)d_in[12];
    float* out = (float*)d_out;

    // workspace layout (bf16 elements)
    __bf16* ws    = (__bf16*)d_ws;
    __bf16* W1aT  = ws;                         // 256*320
    __bf16* W1bT  = W1aT + 256 * 320;           // 256*256
    __bf16* WaT   = W1bT + 256 * 256;           // 8*256*576
    __bf16* WbT   = WaT + 8L * 256 * 576;       // 8*256*256

    wt_transpose_kernel<<<256,  256, 0, stream>>>(W1a, W1aT, 1, IN1,   HID);
    wt_transpose_kernel<<<256,  256, 0, stream>>>(W1b, W1bT, 1, HID,   HID);
    wt_transpose_kernel<<<1024, 256, 0, stream>>>(Wa,  WaT,  8, INMID, HID);
    wt_transpose_kernel<<<1024, 256, 0, stream>>>(Wb,  WbT,  8, HID,   HID);

    const int B = in_sizes[1];           // 131072
    geo2vec_mlp_kernel<<<B / MTILE, 256, 0, stream>>>(
        xy, idx, emb, b1a, b1b, ba, bb, W2, b2, W1aT, W1bT, WaT, WbT, out);
}
